// GraphSageEncoder_78743930404936
// MI455X (gfx1250) — compile-verified, matrix operand order checked by reading, not run on hardware
//
#include <hip/hip_runtime.h>
#include <hip/hip_bf16.h>

// ---------------------------------------------------------------------------
// GraphSAGE encoder for MI455X (gfx1250, wave32, WMMA).
//   layer1: neigh = spmm(A, x);  h1 = relu([x|neigh] @ W1 + b1)
//   layer2: neigh = spmm(A, h1); z  = [h1|neigh] @ Wout + bout
//
// SpMM: device-built CSR (int atomics only) + wave-per-row register-
//       accumulated gather reduce. Feature matrix (102MB) is L2-resident
//       (192MB L2), so gathers run at L2 bandwidth with zero f32 atomics.
// GEMM: V_WMMA_F32_16X16X32_BF16, 4 col-tiles per wave (A-frag reuse),
//       software-pipelined (double-buffered A + 4x B fragments) so loads
//       for k-step ks+1 are in flight while the 4 WMMAs of ks issue.
// ---------------------------------------------------------------------------

#define N_NODES 100000
#define N_EDGES 3200000
#define D       256      // feature dim
#define K_TOT   512      // concat dim (2*D)

typedef __attribute__((ext_vector_type(16))) __bf16 v16bf;
typedef __attribute__((ext_vector_type(8)))  float  v8f;

union FragAB { uint4 u[2]; v16bf v; };

static __device__ __forceinline__ unsigned short f2bf(float f) {
    union { float f; unsigned int u; } x; x.f = f;
    unsigned int r = x.u + 0x7FFFu + ((x.u >> 16) & 1u);   // round-to-nearest-even
    return (unsigned short)(r >> 16);
}

// ---------------------------------------------------------------------------
// W [512][256] f32 (K x N row-major) -> Wt [256][512] bf16 (N x K)
// ---------------------------------------------------------------------------
__global__ void transpose_w_bf16(const float* __restrict__ W,
                                 unsigned short* __restrict__ Wt) {
    int idx = blockIdx.x * blockDim.x + threadIdx.x;   // 0 .. 512*256-1
    int n = idx >> 9;
    int k = idx & 511;
    Wt[idx] = f2bf(W[k * D + n]);
}

// ---------------------------------------------------------------------------
// CSR build: histogram -> exclusive scan -> scatter
// ---------------------------------------------------------------------------
__global__ void count_rows(const int* __restrict__ erow, int* __restrict__ cnt,
                           int nEdges) {
    int e = blockIdx.x * blockDim.x + threadIdx.x;
    if (e < nEdges) atomicAdd(&cnt[erow[e]], 1);
}

// Single-block Hillis-Steele scan over N_NODES counts (runs in ~µs).
__global__ __launch_bounds__(1024)
void scan_counts(const int* __restrict__ cnt, int* __restrict__ rowStart) {
    __shared__ int buf[1024];
    __shared__ int carry;
    if (threadIdx.x == 0) carry = 0;
    __syncthreads();
    for (int base = 0; base < N_NODES; base += 1024) {
        int i = base + threadIdx.x;
        int v = (i < N_NODES) ? cnt[i] : 0;
        int acc = v;
        buf[threadIdx.x] = v;
        __syncthreads();
        for (int off = 1; off < 1024; off <<= 1) {
            int t = (threadIdx.x >= off) ? buf[threadIdx.x - off] : 0;
            __syncthreads();
            acc += t;
            buf[threadIdx.x] = acc;
            __syncthreads();
        }
        if (i < N_NODES) rowStart[i] = carry + acc - v;   // exclusive
        __syncthreads();
        if (threadIdx.x == 1023) carry += buf[1023];
        __syncthreads();
    }
    if (threadIdx.x == 0) rowStart[N_NODES] = carry;      // == N_EDGES
}

__global__ void scatter_edges(const int* __restrict__ erow, const int* __restrict__ ecol,
                              const float* __restrict__ eval, int* __restrict__ cursor,
                              int* __restrict__ ecolS, float* __restrict__ evalS,
                              int nEdges) {
    int e = blockIdx.x * blockDim.x + threadIdx.x;
    if (e >= nEdges) return;
    int r   = erow[e];
    int pos = atomicAdd(&cursor[r], 1);
    ecolS[pos] = ecol[e];
    evalS[pos] = eval[e];
}

// ---------------------------------------------------------------------------
// SpMM (CSR): one wave per row. 256 floats = 2x float4 per lane held in
// registers. Edge (col,val) pairs loaded coalesced 32-at-a-time, broadcast
// with __shfl. No float atomics, no output pre-zeroing needed.
// ---------------------------------------------------------------------------
__global__ __launch_bounds__(256)
void spmm_csr(const int* __restrict__ rowStart, const int* __restrict__ ecolS,
              const float* __restrict__ evalS, const float* __restrict__ h,
              float* __restrict__ out) {
    int row  = (int)((blockIdx.x * (unsigned)blockDim.x + threadIdx.x) >> 5);
    int lane = threadIdx.x & 31;
    if (row >= N_NODES) return;

    int s   = rowStart[row];
    int end = rowStart[row + 1];

    float4 acc0 = {0.f, 0.f, 0.f, 0.f};
    float4 acc1 = {0.f, 0.f, 0.f, 0.f};

    for (int base = s; base < end; base += 32) {
        int cnt = end - base;
        if (cnt > 32) cnt = 32;
        int   cc = 0;
        float vv = 0.0f;
        if (lane < cnt) {
            cc = ecolS[base + lane];
            vv = evalS[base + lane];
        }
        for (int j = 0; j < cnt; ++j) {
            float vj = __shfl(vv, j);
            int   cj = __shfl(cc, j);
            const float4* hp = (const float4*)(h + (size_t)cj * D);
            float4 a = hp[lane];
            float4 b = hp[lane + 32];
            acc0.x += vj * a.x; acc0.y += vj * a.y;
            acc0.z += vj * a.z; acc0.w += vj * a.w;
            acc1.x += vj * b.x; acc1.y += vj * b.y;
            acc1.z += vj * b.z; acc1.w += vj * b.w;
        }
    }

    float4* o = (float4*)(out + (size_t)row * D);
    o[lane]      = acc0;
    o[lane + 32] = acc1;
}

// ---------------------------------------------------------------------------
// Fused concat + GEMM + bias (+ ReLU):
//   out[m, n] = act( sum_k [hSelf|hNeigh][m,k] * Wt[n,k] + bias[n] )
//
// Block: 256 threads = 8 waves in a 2x4 grid over a 32-row x 256-col tile.
// Each wave: 1 row-tile x 4 col-tiles, K = 512 = 16 k-steps of 32.
// Double-buffered fragments: loads for ks+1 issue before the WMMAs of ks.
//
// bf16 16x16x32 fragment layout (ISA 7.12.2): lane L -> m/n = L&15,
// half = L>>4; K = 16*g + 8*half + (0..7) => two 16B runs per lane/k-step.
// LDS row stride 520 shorts (1040B): each half-wave's 16B reads cover all
// 64 banks exactly once -> conflict-free.
// ---------------------------------------------------------------------------
template <bool RELU>
__global__ __launch_bounds__(256)
void sage_wmma_gemm(const float* __restrict__ hSelf,
                    const float* __restrict__ hNeigh,
                    const unsigned short* __restrict__ Wt,   // [256][512] bf16
                    const float* __restrict__ bias,
                    float* __restrict__ out) {
    __shared__ alignas(16) unsigned short A_lds[32][520];

    const int    tid     = threadIdx.x;
    const size_t rowBase = (size_t)blockIdx.x * 32;

    // ---- stage A tile (32 rows x 512 K) as bf16: concat [hSelf | hNeigh] ----
    #pragma unroll
    for (int i = 0; i < 64; ++i) {
        int linear = tid + i * 256;      // 0 .. 16383
        int m = linear >> 9;
        int k = linear & 511;
        size_t node = rowBase + (size_t)m;
        float f = (k < D) ? hSelf[node * D + k] : hNeigh[node * D + (k - D)];
        A_lds[m][k] = f2bf(f);
    }
    __syncthreads();

    const int wave    = tid >> 5;
    const int lane    = tid & 31;
    const int waveRow = wave >> 2;                    // 0..1 -> 16-row tile
    const int waveCol = wave & 3;                     // 0..3 -> 64-col group
    const int m       = (waveRow << 4) + (lane & 15); // A fragment row in LDS
    const int half    = lane >> 4;                    // K sub-block select
    const int n0      = waveCol * 64 + (lane & 15);   // first of 4 col tiles

    v8f acc[4];
    #pragma unroll
    for (int t = 0; t < 4; ++t) {
        float bv = bias[n0 + t * 16];
        v8f z = { bv, bv, bv, bv, bv, bv, bv, bv };
        acc[t] = z;
    }

    // Per-lane base pointers; all remaining offsets are compile-time
    // immediates in the unrolled loop.
    const unsigned short* wbase = Wt + (size_t)n0 * K_TOT + half * 8;
    const unsigned short* abase = &A_lds[m][half * 8];

    FragAB fa[2];
    FragAB fb[2][4];

    // Prologue: fetch k-step 0.
    fa[0].u[0] = *(const uint4*)(abase);
    fa[0].u[1] = *(const uint4*)(abase + 16);
    #pragma unroll
    for (int t = 0; t < 4; ++t) {
        fb[0][t].u[0] = *(const uint4*)(wbase + (size_t)t * 16 * K_TOT);
        fb[0][t].u[1] = *(const uint4*)(wbase + (size_t)t * 16 * K_TOT + 16);
    }

    #pragma unroll
    for (int ks = 0; ks < 16; ++ks) {
        const int cur = ks & 1;
        const int nxt = cur ^ 1;
        if (ks < 15) {
            const int kb = (ks + 1) * 32;                 // next k-step offset
            fa[nxt].u[0] = *(const uint4*)(abase + kb);
            fa[nxt].u[1] = *(const uint4*)(abase + kb + 16);
            #pragma unroll
            for (int t = 0; t < 4; ++t) {
                fb[nxt][t].u[0] = *(const uint4*)(wbase + (size_t)t * 16 * K_TOT + kb);
                fb[nxt][t].u[1] = *(const uint4*)(wbase + (size_t)t * 16 * K_TOT + kb + 16);
            }
        }
        #pragma unroll
        for (int t = 0; t < 4; ++t) {
            acc[t] = __builtin_amdgcn_wmma_f32_16x16x32_bf16(
                         /*neg_a=*/false, fa[cur].v, /*neg_b=*/false, fb[cur][t].v,
                         /*c_mod=*/(short)0, acc[t],
                         /*reuse_a=*/false, /*reuse_b=*/false);
        }
    }

    // C/D layout: lane L, reg r -> (m = r + 8*half, n = L&15)
    #pragma unroll
    for (int t = 0; t < 4; ++t) {
        float* optr = out + (rowBase + (size_t)(waveRow * 16 + half * 8)) * D
                          + n0 + t * 16;
        #pragma unroll
        for (int r = 0; r < 8; ++r) {
            float v = acc[t][r];
            if (RELU) v = v > 0.0f ? v : 0.0f;
            optr[(size_t)r * D] = v;
        }
    }
}

// ---------------------------------------------------------------------------
// Host-side orchestration (graph-capture safe: only async ops on `stream`).
// ---------------------------------------------------------------------------
extern "C" void kernel_launch(void* const* d_in, const int* in_sizes, int n_in,
                              void* d_out, int out_size, void* d_ws, size_t ws_size,
                              hipStream_t stream) {
    const float* x    = (const float*)d_in[0];
    const int*   erow = (const int*)  d_in[1];
    const int*   ecol = (const int*)  d_in[2];
    const float* eval = (const float*)d_in[3];
    const float* W1   = (const float*)d_in[4];
    const float* b1   = (const float*)d_in[5];
    const float* Wout = (const float*)d_in[6];
    const float* bout = (const float*)d_in[7];
    float* out = (float*)d_out;

    const size_t featBytes = (size_t)N_NODES * D * sizeof(float);  // 102.4 MB
    char* ws = (char*)d_ws;
    size_t off = 0;
    float*          neigh   = (float*)(ws + off);           off += featBytes;
    float*          h1      = (float*)(ws + off);           off += featBytes;
    unsigned short* W1t     = (unsigned short*)(ws + off);  off += (size_t)K_TOT * D * 2;
    unsigned short* Woutt   = (unsigned short*)(ws + off);  off += (size_t)K_TOT * D * 2;
    int*            cnt     = (int*)(ws + off);             off += (size_t)N_NODES * 4;
    int*            rowStart= (int*)(ws + off);             off += (size_t)(N_NODES + 4) * 4;
    int*            cursor  = (int*)(ws + off);             off += (size_t)N_NODES * 4;
    int*            ecolS   = (int*)(ws + off);             off += (size_t)N_EDGES * 4;
    float*          evalS   = (float*)(ws + off);           off += (size_t)N_EDGES * 4;

    // Weights -> bf16, transposed to [N][K]
    transpose_w_bf16<<<(K_TOT * D) / 256, 256, 0, stream>>>(W1,   W1t);
    transpose_w_bf16<<<(K_TOT * D) / 256, 256, 0, stream>>>(Wout, Woutt);

    // CSR build (once per launch; reused by both layers)
    const int edgeBlocks = (N_EDGES + 255) / 256;   // 12500
    hipMemsetAsync(cnt, 0, (size_t)N_NODES * 4, stream);
    count_rows<<<edgeBlocks, 256, 0, stream>>>(erow, cnt, N_EDGES);
    scan_counts<<<1, 1024, 0, stream>>>(cnt, rowStart);
    hipMemcpyAsync(cursor, rowStart, (size_t)N_NODES * 4,
                   hipMemcpyDeviceToDevice, stream);
    scatter_edges<<<edgeBlocks, 256, 0, stream>>>(erow, ecol, eval, cursor,
                                                  ecolS, evalS, N_EDGES);

    const int spmmBlocks = (N_NODES * 32 + 255) / 256;  // wave per row: 12500
    const int gemmBlocks = N_NODES / 32;                // 3125 (exact)

    // Layer 1
    spmm_csr<<<spmmBlocks, 256, 0, stream>>>(rowStart, ecolS, evalS, x, neigh);
    sage_wmma_gemm<true><<<gemmBlocks, 256, 0, stream>>>(x, neigh, W1t, b1, h1);

    // Layer 2
    spmm_csr<<<spmmBlocks, 256, 0, stream>>>(rowStart, ecolS, evalS, h1, neigh);
    sage_wmma_gemm<false><<<gemmBlocks, 256, 0, stream>>>(h1, neigh, Woutt, bout, out);
}
